// DisentangleMultiHeadedAttention_23347442221587
// MI455X (gfx1250) — matrix operand order repaired and checked by reading, not compiled, not run on hardware
//
#include <hip/hip_runtime.h>
#include <hip/hip_bf16.h>

// ---------------------------------------------------------------- constants
#define BATCH 4
#define SEQ   1024
#define HID   1024
#define HEADS 16
#define DH    64

typedef __attribute__((ext_vector_type(16))) __bf16        v16bf;
typedef __attribute__((ext_vector_type(8)))  float         v8f;
typedef __attribute__((ext_vector_type(4)))  unsigned int  v4u;
typedef __attribute__((ext_vector_type(4)))  int           v4i;

union FragU { v4u q[2]; v16bf bf; };

#define AS1 __attribute__((address_space(1)))
#define AS3 __attribute__((address_space(3)))

#if __has_builtin(__builtin_amdgcn_global_load_async_to_lds_b128) && \
    __has_builtin(__builtin_amdgcn_s_wait_asynccnt)
#define HAVE_ASYNC_LDS 1
#else
#define HAVE_ASYNC_LDS 0
#endif

// 16B global -> LDS copy. Async (ASYNCcnt) when available, else sync fallback.
__device__ inline void copy16_g2l(const void* g, void* l) {
#if HAVE_ASYNC_LDS
    __builtin_amdgcn_global_load_async_to_lds_b128(
        (AS1 v4i*)(uintptr_t)g,
        (AS3 v4i*)(unsigned int)(uintptr_t)l, 0, 0);
#else
    *(v4u*)l = *(const v4u*)g;
#endif
}
__device__ inline void async_wait() {
#if HAVE_ASYNC_LDS
    __builtin_amdgcn_s_wait_asynccnt(0);
#endif
}

// ---- 16-lane-row butterfly reductions: DPP (VALU) instead of ds_bpermute
#if __has_builtin(__builtin_amdgcn_update_dpp)
template<int CTRL>
__device__ inline float dpp_swap(float x) {
    return __int_as_float(__builtin_amdgcn_update_dpp(
        0, __float_as_int(x), CTRL, 0xf, 0xf, true));
}
__device__ inline float rowmax16(float x) {
    x = fmaxf(x, dpp_swap<0xB1>(x));    // quad_perm(1,0,3,2): xor 1
    x = fmaxf(x, dpp_swap<0x4E>(x));    // quad_perm(2,3,0,1): xor 2
    x = fmaxf(x, dpp_swap<0x164>(x));   // row_xmask:4
    x = fmaxf(x, dpp_swap<0x168>(x));   // row_xmask:8
    return x;
}
__device__ inline float rowsum16(float x) {
    x += dpp_swap<0xB1>(x);
    x += dpp_swap<0x4E>(x);
    x += dpp_swap<0x164>(x);
    x += dpp_swap<0x168>(x);
    return x;
}
#else
__device__ inline float rowmax16(float x) {
    x = fmaxf(x, __shfl_xor(x, 1)); x = fmaxf(x, __shfl_xor(x, 2));
    x = fmaxf(x, __shfl_xor(x, 4)); x = fmaxf(x, __shfl_xor(x, 8));
    return x;
}
__device__ inline float rowsum16(float x) {
    x += __shfl_xor(x, 1); x += __shfl_xor(x, 2);
    x += __shfl_xor(x, 4); x += __shfl_xor(x, 8);
    return x;
}
#endif

__device__ inline unsigned short f2bf(float f) {
    unsigned int u = __float_as_uint(f);
    u += 0x7fffu + ((u >> 16) & 1u);          // round-to-nearest-even
    return (unsigned short)(u >> 16);
}
__device__ inline float bf2f(unsigned short h) {
    return __uint_as_float(((unsigned int)h) << 16);
}

__device__ inline v8f wmma_bf16(v16bf a, v16bf b, v8f c) {
    return __builtin_amdgcn_wmma_f32_16x16x32_bf16(
        false, a, false, b, (short)0, c, false, false);
}

// A fragment (16x32, M x K): lanes 0-15 -> M=lane, K={0..7,16..23};
// lanes 16-31 -> M=lane-16, K={8..15,24..31}. Two ds_load_b128 per lane.
__device__ inline v16bf load_fragA(const unsigned short* lds, int stride, int lane) {
    int m  = lane & 15;
    int k0 = (lane & 16) ? 8 : 0;
    const unsigned short* p = lds + m * stride + k0;
    FragU f;
    f.q[0] = *(const v4u*)(p);
    f.q[1] = *(const v4u*)(p + 16);
    return f.bf;
}

// B fragment (32x16, K x N) from an LDS tile stored [n][k] (k-contiguous):
// lanes 0-15 -> N=lane, K={0..7,16..23}; lanes 16-31 -> N=lane-16, K={8..15,24..31}.
__device__ inline v16bf load_fragB(const unsigned short* lds, int stride, int lane) {
    int n  = lane & 15;
    int k0 = (lane & 16) ? 8 : 0;
    const unsigned short* p = lds + n * stride + k0;
    FragU f;
    f.q[0] = *(const v4u*)(p);
    f.q[1] = *(const v4u*)(p + 16);
    return f.bf;
}

// ------------------------------------------------- fp32 -> bf16 conversion
__global__ __launch_bounds__(256)
void cvt_bf16(const float* __restrict__ in, unsigned short* __restrict__ out, int n4)
{
    int i = blockIdx.x * 256 + threadIdx.x;
    if (i < n4) {
        float4 v = ((const float4*)in)[i];
        unsigned int lo = (unsigned)f2bf(v.x) | ((unsigned)f2bf(v.y) << 16);
        unsigned int hi = (unsigned)f2bf(v.z) | ((unsigned)f2bf(v.w) << 16);
        ((uint2*)out)[i] = make_uint2(lo, hi);
    }
}

// ---------------------------------------------------------------- GEMM (NT)
// C[M,N] = A[M,K] @ W[N,K]^T + bias.  bf16 operands, double-buffered async
// LDS staging.  Block tile 64x128, 8 waves (2m x 4n), 4 WMMA/wave/step.
#define TK   32
#define LSTR 40   // bf16 elements; 80B row stride: 16B aligned, conflict-free

template<bool OUTF32>
__global__ __launch_bounds__(256)
void gemm_nt(const unsigned short* __restrict__ A,   // [M x K] bf16
             const unsigned short* __restrict__ W,   // [N x K] bf16
             const float* __restrict__ bias,
             unsigned short* __restrict__ Cbf, float* __restrict__ Cf,
             int Kdim, int Ndim)
{
    __shared__ __align__(16) unsigned short sA[2][ 64 * LSTR];
    __shared__ __align__(16) unsigned short sB[2][128 * LSTR];

    const int tid = threadIdx.x, lane = tid & 31, wid = tid >> 5;
    const int m0 = blockIdx.x * 64, n0 = blockIdx.y * 128;
    const int wn = wid & 3, wm = wid >> 2;           // 2(m) x 4(n) wave grid

    v8f c00 = {}, c01 = {}, c10 = {}, c11 = {};

    // per-thread staging chunks (16B = 8 bf16 each)
    const int ra  = tid >> 2,         ca  = (tid & 3) * 8;          // A: 256 chunks
    const int rb0 = tid >> 2,         cb0 = (tid & 3) * 8;          // B: 512 chunks
    const int rb1 = (tid + 256) >> 2, cb1 = ((tid + 256) & 3) * 8;

    const size_t rowA  = (size_t)(m0 + ra)  * Kdim + ca;
    const size_t rowB0 = (size_t)(n0 + rb0) * Kdim + cb0;
    const size_t rowB1 = (size_t)(n0 + rb1) * Kdim + cb1;

    // prologue: stage k0 = 0 into buffer 0
    copy16_g2l(A + rowA,  &sA[0][ra  * LSTR + ca]);
    copy16_g2l(W + rowB0, &sB[0][rb0 * LSTR + cb0]);
    copy16_g2l(W + rowB1, &sB[0][rb1 * LSTR + cb1]);
    async_wait();
    __syncthreads();

    int buf = 0;
    for (int k0 = 0; k0 < Kdim; k0 += TK) {
        const int kn = k0 + TK;
        if (kn < Kdim) {            // prefetch next K-tile into other buffer
            copy16_g2l(A + rowA  + kn, &sA[buf ^ 1][ra  * LSTR + ca]);
            copy16_g2l(W + rowB0 + kn, &sB[buf ^ 1][rb0 * LSTR + cb0]);
            copy16_g2l(W + rowB1 + kn, &sB[buf ^ 1][rb1 * LSTR + cb1]);
        }
        v16bf a0 = load_fragA(&sA[buf][(wm * 32)      * LSTR], LSTR, lane);
        v16bf a1 = load_fragA(&sA[buf][(wm * 32 + 16) * LSTR], LSTR, lane);
        v16bf b0 = load_fragB(&sB[buf][(wn * 32)      * LSTR], LSTR, lane);
        v16bf b1 = load_fragB(&sB[buf][(wn * 32 + 16) * LSTR], LSTR, lane);
        c00 = wmma_bf16(a0, b0, c00);
        c01 = wmma_bf16(a0, b1, c01);
        c10 = wmma_bf16(a1, b0, c10);
        c11 = wmma_bf16(a1, b1, c11);
        async_wait();               // own prefetch complete
        __syncthreads();            // everyone done reading buf / writing buf^1
        buf ^= 1;
    }

    // C fragment: VGPR r, lanes 0-15 -> row r, lanes 16-31 -> row r+8; col = lane&15
    const int rowb = m0 + wm * 32 + ((lane & 16) ? 8 : 0);
    const int colA = n0 + wn * 32 + (lane & 15);
    const int colB = colA + 16;
    const float bA = bias[colA], bB = bias[colB];
    #pragma unroll
    for (int r = 0; r < 8; ++r) {
        size_t r0 = (size_t)(rowb + r)      * Ndim;
        size_t r1 = (size_t)(rowb + 16 + r) * Ndim;
        float x00 = c00[r] + bA, x01 = c01[r] + bB;
        float x10 = c10[r] + bA, x11 = c11[r] + bB;
        if constexpr (OUTF32) {
            Cf[r0 + colA] = x00;  Cf[r0 + colB] = x01;
            Cf[r1 + colA] = x10;  Cf[r1 + colB] = x11;
        } else {
            Cbf[r0 + colA] = f2bf(x00);  Cbf[r0 + colB] = f2bf(x01);
            Cbf[r1 + colA] = f2bf(x10);  Cbf[r1 + colB] = f2bf(x11);
        }
    }
}

// ---------------------------------------------------------- ksum = ka + kb
__global__ __launch_bounds__(256)
void add_bf16(const unsigned short* __restrict__ a,
              const unsigned short* __restrict__ b,
              unsigned short* __restrict__ c, int n)
{
    int i = blockIdx.x * blockDim.x + threadIdx.x;
    if (i < n) c[i] = f2bf(bf2f(a[i]) + bf2f(b[i]));
}

// ------------------------------------------------------- flash attention
// grid: (B*HEADS, SEQ/128); 256 threads = 8 waves, each wave owns 16 q rows.
// Double-buffered K/V staging (async K copies; V transposed via registers).
#define AQ_STRIDE 72
#define AK_STRIDE 72
#define AV_STRIDE 40
#define AP_STRIDE 40

__global__ __launch_bounds__(256)
void attn_flash(const unsigned short* __restrict__ Q,
                const unsigned short* __restrict__ Ks,
                const unsigned short* __restrict__ V,
                const int* __restrict__ mask,
                unsigned short* __restrict__ O, float sl2e)
{
    __shared__ __align__(16) unsigned short sQ[128 * AQ_STRIDE];
    __shared__ __align__(16) unsigned short sK[2][32 * AK_STRIDE];
    __shared__ __align__(16) unsigned short sVt[2][DH * AV_STRIDE];  // transposed
    __shared__ __align__(16) unsigned short sP[8 * 16 * AP_STRIDE];  // per-wave

    const int tid = threadIdx.x, lane = tid & 31, wid = tid >> 5;
    const int bh = blockIdx.x, b = bh >> 4, h = bh & 15;
    const int q0 = blockIdx.y * 128;
    const size_t base = ((size_t)b * SEQ) * HID + (size_t)h * DH;

    // stage all waves' Q (128 x 64 bf16) once: 1024 16B chunks, async
    #pragma unroll
    for (int j = 0; j < 4; ++j) {
        int i = tid + j * 256;
        int r = i >> 3, c8 = (i & 7) * 8;
        copy16_g2l(Q + base + (size_t)(q0 + r) * HID + c8, &sQ[r * AQ_STRIDE + c8]);
    }

    // prologue: stage key-chunk 0 into buffer 0
    const int rk = tid >> 3, ck = (tid & 7) * 8;     // K chunk per thread
    copy16_g2l(Ks + base + (size_t)rk * HID + ck, &sK[0][rk * AK_STRIDE + ck]);
    for (int i = tid; i < 32 * 32; i += 256) {
        int r = i >> 5, c2 = (i & 31) * 2;
        unsigned int vv = *(const unsigned int*)(V + base + (size_t)r * HID + c2);
        sVt[0][c2 * AV_STRIDE + r]       = (unsigned short)(vv & 0xffffu);
        sVt[0][(c2 + 1) * AV_STRIDE + r] = (unsigned short)(vv >> 16);
    }
    async_wait();
    __syncthreads();

    const v16bf aq0 = load_fragA(&sQ[(wid * 16) * AQ_STRIDE],      AQ_STRIDE, lane);
    const v16bf aq1 = load_fragA(&sQ[(wid * 16) * AQ_STRIDE + 32], AQ_STRIDE, lane);

    float m_i[8], l_i[8];
    #pragma unroll
    for (int r = 0; r < 8; ++r) { m_i[r] = -3.0e38f; l_i[r] = 0.f; }
    v8f o0 = {}, o1 = {}, o2 = {}, o3 = {};

    const int kidx = lane & 15;
    const int rowadd = (lane & 16) ? 8 : 0;
    unsigned short* pP = &sP[wid * 16 * AP_STRIDE + kidx];

    int buf = 0;
    for (int kc = 0; kc < SEQ; kc += 32) {
        const int kn = kc + 32;
        const bool pre = kn < SEQ;
        // ---- prefetch next chunk: async K; V fetched to registers early so
        //      the loadcnt wait lands after the score WMMAs.
        unsigned int vv[4] = {0u, 0u, 0u, 0u};
        if (pre) {
            copy16_g2l(Ks + base + (size_t)(kn + rk) * HID + ck,
                       &sK[buf ^ 1][rk * AK_STRIDE + ck]);
            #pragma unroll
            for (int j = 0; j < 4; ++j) {
                int i = tid + j * 256;
                int r = i >> 5, c2 = (i & 31) * 2;
                vv[j] = *(const unsigned int*)(V + base + (size_t)(kn + r) * HID + c2);
            }
        }

        // ---- scores for both 16-key subtiles (K-dim = DH = 64 -> 2 wmma each)
        v8f s0 = {}, s1 = {};
        {
            v16bf bk0 = load_fragB(&sK[buf][0],                   AK_STRIDE, lane);
            v16bf bk1 = load_fragB(&sK[buf][32],                  AK_STRIDE, lane);
            s0 = wmma_bf16(aq0, bk0, s0);
            s0 = wmma_bf16(aq1, bk1, s0);
            v16bf bk2 = load_fragB(&sK[buf][16 * AK_STRIDE],      AK_STRIDE, lane);
            v16bf bk3 = load_fragB(&sK[buf][16 * AK_STRIDE + 32], AK_STRIDE, lane);
            s1 = wmma_bf16(aq0, bk2, s1);
            s1 = wmma_bf16(aq1, bk3, s1);
        }

        const bool ok0 = mask[b * SEQ + kc + kidx]      != 0;
        const bool ok1 = mask[b * SEQ + kc + 16 + kidx] != 0;

        // ---- online softmax (exp2 domain); DPP row reductions (pure VALU)
        #pragma unroll
        for (int r = 0; r < 8; ++r) {
            float x0 = ok0 ? s0[r] * sl2e : -3.0e38f;
            float x1 = ok1 ? s1[r] * sl2e : -3.0e38f;
            float rm = rowmax16(fmaxf(x0, x1));
            float mnew  = fmaxf(m_i[r], rm);
            float alpha = __builtin_amdgcn_exp2f(m_i[r] - mnew);
            float p0    = __builtin_amdgcn_exp2f(x0 - mnew);
            float p1    = __builtin_amdgcn_exp2f(x1 - mnew);
            float ps    = rowsum16(p0 + p1);
            l_i[r] = l_i[r] * alpha + ps;
            m_i[r] = mnew;
            o0[r] *= alpha; o1[r] *= alpha; o2[r] *= alpha; o3[r] *= alpha;
            pP[(r + rowadd) * AP_STRIDE]      = f2bf(p0);   // C-layout -> A-layout
            pP[(r + rowadd) * AP_STRIDE + 16] = f2bf(p1);   // via per-wave LDS
        }

        // ---- store transposed V of next chunk (into other buffer) while the
        //      current PV runs on sVt[buf]
        if (pre) {
            #pragma unroll
            for (int j = 0; j < 4; ++j) {
                int i = tid + j * 256;
                int r = i >> 5, c2 = (i & 31) * 2;
                sVt[buf ^ 1][c2 * AV_STRIDE + r]       = (unsigned short)(vv[j] & 0xffffu);
                sVt[buf ^ 1][(c2 + 1) * AV_STRIDE + r] = (unsigned short)(vv[j] >> 16);
            }
        }

        // ---- P(16x32) @ V(32x64); per-wave DS ops are in-order
        v16bf ap  = load_fragA(&sP[wid * 16 * AP_STRIDE], AP_STRIDE, lane);
        v16bf bv0 = load_fragB(&sVt[buf][0],              AV_STRIDE, lane);
        v16bf bv1 = load_fragB(&sVt[buf][16 * AV_STRIDE], AV_STRIDE, lane);
        v16bf bv2 = load_fragB(&sVt[buf][32 * AV_STRIDE], AV_STRIDE, lane);
        v16bf bv3 = load_fragB(&sVt[buf][48 * AV_STRIDE], AV_STRIDE, lane);
        o0 = wmma_bf16(ap, bv0, o0);
        o1 = wmma_bf16(ap, bv1, o1);
        o2 = wmma_bf16(ap, bv2, o2);
        o3 = wmma_bf16(ap, bv3, o3);

        async_wait();               // own async K prefetch complete
        __syncthreads();            // all waves done with buf / done writing buf^1
        buf ^= 1;
    }

    // normalize + store bf16 [b, q, h*64 + d] (feeds output-projection GEMM)
    #pragma unroll
    for (int r = 0; r < 8; ++r) {
        float inv = 1.0f / l_i[r];
        int row = q0 + wid * 16 + r + rowadd;
        size_t orow = base + (size_t)row * HID;
        int d = kidx;
        O[orow + d]      = f2bf(o0[r] * inv);
        O[orow + 16 + d] = f2bf(o1[r] * inv);
        O[orow + 32 + d] = f2bf(o2[r] * inv);
        O[orow + 48 + d] = f2bf(o3[r] * inv);
    }
}

// ---------------------------------------------------------------- launcher
extern "C" void kernel_launch(void* const* d_in, const int* in_sizes, int n_in,
                              void* d_out, int out_size, void* d_ws, size_t ws_size,
                              hipStream_t stream)
{
    const float* xin[6] = { (const float*)d_in[0], (const float*)d_in[1],
                            (const float*)d_in[2], (const float*)d_in[3],
                            (const float*)d_in[4], (const float*)d_in[5] };
    const int*   mask = (const int*)  d_in[6];
    const float* Wa   = (const float*)d_in[7];
    const float* ba   = (const float*)d_in[8];
    const float* Wb   = (const float*)d_in[9];
    const float* bb   = (const float*)d_in[10];
    const float* Wo_a = (const float*)d_in[11];
    const float* bo_a = (const float*)d_in[12];
    const float* Wo_b = (const float*)d_in[13];
    const float* bo_b = (const float*)d_in[14];
    float* out = (float*)d_out;

    const size_t PE = (size_t)BATCH * SEQ * HID;     // 4M activation elems
    const size_t WW = (size_t)HID * HID;             // 1M weight elems

    // ws layout (bf16 elements): 6 inputs | 8M weights | 7 proj | 2 attn-out
    unsigned short* w16 = (unsigned short*)d_ws;
    unsigned short* xin16[6];
    for (int i = 0; i < 6; ++i) xin16[i] = w16 + i * PE;
    unsigned short* wa16  = w16 + 6 * PE;            // 3*WW
    unsigned short* wb16  = wa16 + 3 * WW;           // 3*WW
    unsigned short* woa16 = wb16 + 3 * WW;           // WW
    unsigned short* wob16 = woa16 + WW;              // WW
    unsigned short* prj   = wob16 + WW;
    unsigned short* qa    = prj + 0 * PE;
    unsigned short* ka    = prj + 1 * PE;
    unsigned short* va    = prj + 2 * PE;
    unsigned short* qb    = prj + 3 * PE;
    unsigned short* kb    = prj + 4 * PE;
    unsigned short* vb    = prj + 5 * PE;
    unsigned short* ksum  = prj + 6 * PE;
    unsigned short* ha    = prj + 7 * PE;
    unsigned short* hb    = prj + 8 * PE;            // end = 136 MiB

    dim3 blk(256);

    // fp32 -> bf16 conversion passes (bandwidth-bound, ~200 MB total)
    for (int i = 0; i < 6; ++i)
        cvt_bf16<<<dim3((unsigned)(PE / 4 / 256)), blk, 0, stream>>>(xin[i], xin16[i], (int)(PE / 4));
    cvt_bf16<<<dim3((unsigned)(3 * WW / 4 / 256)), blk, 0, stream>>>(Wa,   wa16,  (int)(3 * WW / 4));
    cvt_bf16<<<dim3((unsigned)(3 * WW / 4 / 256)), blk, 0, stream>>>(Wb,   wb16,  (int)(3 * WW / 4));
    cvt_bf16<<<dim3((unsigned)(WW / 4 / 256)),     blk, 0, stream>>>(Wo_a, woa16, (int)(WW / 4));
    cvt_bf16<<<dim3((unsigned)(WW / 4 / 256)),     blk, 0, stream>>>(Wo_b, wob16, (int)(WW / 4));

    dim3 gg(BATCH * SEQ / 64, HID / 128);            // 64 x 8

    // 6 input projections  (x @ W^T + b) -> bf16
    gemm_nt<false><<<gg, blk, 0, stream>>>(xin16[0], wa16 + 0 * WW, ba + 0 * HID, qa, nullptr, HID, HID);
    gemm_nt<false><<<gg, blk, 0, stream>>>(xin16[1], wa16 + 1 * WW, ba + 1 * HID, ka, nullptr, HID, HID);
    gemm_nt<false><<<gg, blk, 0, stream>>>(xin16[2], wa16 + 2 * WW, ba + 2 * HID, va, nullptr, HID, HID);
    gemm_nt<false><<<gg, blk, 0, stream>>>(xin16[3], wb16 + 0 * WW, bb + 0 * HID, qb, nullptr, HID, HID);
    gemm_nt<false><<<gg, blk, 0, stream>>>(xin16[4], wb16 + 1 * WW, bb + 1 * HID, kb, nullptr, HID, HID);
    gemm_nt<false><<<gg, blk, 0, stream>>>(xin16[5], wb16 + 2 * WW, bb + 2 * HID, vb, nullptr, HID, HID);

    // ksum = ka + kb  (both streams attend against qa@(ka+kb)^T / qb@(ka+kb)^T)
    add_bf16<<<dim3((unsigned)((PE + 255) / 256)), blk, 0, stream>>>(ka, kb, ksum, (int)PE);

    // flash attention per stream
    const float scale = 0.08838834764831845f;        // 1/sqrt(2*DH) = 1/sqrt(128)
    const float sl2e  = scale * 1.4426950408889634f; // fold log2(e) for exp2
    dim3 ga(BATCH * HEADS, SEQ / 128);               // 64 x 8
    attn_flash<<<ga, blk, 0, stream>>>(qa, ksum, va, mask, ha, sl2e);
    attn_flash<<<ga, blk, 0, stream>>>(qb, ksum, vb, mask, hb, sl2e);

    // output projections -> fp32 d_out (stream a then stream b)
    gemm_nt<true><<<gg, blk, 0, stream>>>(ha, woa16, bo_a, nullptr, out,      HID, HID);
    gemm_nt<true><<<gg, blk, 0, stream>>>(hb, wob16, bo_b, nullptr, out + PE, HID, HID);
}
